// MyModel_87522843560452
// MI455X (gfx1250) — matrix operand order repaired and checked by reading, not compiled, not run on hardware
//
#include <hip/hip_runtime.h>
#include <hip/hip_bf16.h>

typedef __attribute__((ext_vector_type(16))) _Float16 v16h;
typedef __attribute__((ext_vector_type(8)))  _Float16 v8h;
typedef __attribute__((ext_vector_type(8)))  float    v8f;

#define T_SEQ  512
#define BATCH  256
#define F_IN   128

// Layer 1: H=128, gates=512, K = F_IN + H = 256
#define H1     128
#define G1     512
#define K1     256
#define WP1    264      // K1 + 8 halves pad (528B rows -> bank spread)
#define AP1    136      // A-tile pitch in halves (272B, 16B aligned)

// Layer 2: H=64, gates=256, K = 256 + 64 = 320
#define H2     64
#define G2     256
#define K2     320
#define WP2    328      // K2 + 8 halves pad
#define AP2    72       // h-tile pitch (144B, 16B aligned)

__device__ __forceinline__ float sigmoidf_(float x) {
  return 1.0f / (1.0f + __expf(-x));
}

// ---------------------------------------------------------------------------
// Embedding MLP branch: x = relu(relu(relu(emb[A].flat @ w1+b1) @ w2+b2) @ w3+b3)
// Tiny (~0.1 GFLOP): one block per batch row.
// ---------------------------------------------------------------------------
__global__ __launch_bounds__(256) void emb_mlp_kernel(
    const int* __restrict__ inputA, const float* __restrict__ emb,
    const float* __restrict__ w1, const float* __restrict__ b1,
    const float* __restrict__ w2, const float* __restrict__ b2,
    const float* __restrict__ w3, const float* __restrict__ b3,
    float* __restrict__ x64)
{
  __shared__ float xe[600];
  __shared__ float h1[256];
  __shared__ float h2[128];
  const int b = blockIdx.x;
  const int tid = threadIdx.x;
  if (tid < 200) {
    int tok = inputA[b * 200 + tid];
    xe[tid * 3 + 0] = emb[tok * 3 + 0];
    xe[tid * 3 + 1] = emb[tok * 3 + 1];
    xe[tid * 3 + 2] = emb[tok * 3 + 2];
  }
  __syncthreads();
  {
    float s = b1[tid];
    for (int k = 0; k < 600; ++k) s += xe[k] * w1[k * 256 + tid];
    h1[tid] = s > 0.f ? s : 0.f;
  }
  __syncthreads();
  if (tid < 128) {
    float s = b2[tid];
    for (int k = 0; k < 256; ++k) s += h1[k] * w2[k * 128 + tid];
    h2[tid] = s > 0.f ? s : 0.f;
  }
  __syncthreads();
  if (tid < 64) {
    float s = b3[tid];
    for (int k = 0; k < 128; ++k) s += h2[k] * w3[k * 64 + tid];
    x64[b * 64 + tid] = s > 0.f ? s : 0.f;
  }
}

// ---------------------------------------------------------------------------
// Layer-1 LSTM scan (both directions via blockIdx.y). 16 batch rows / block,
// 8 waves, wave w owns hidden columns [16w, 16w+16). Combined transposed
// weights [Wk;Wr] as f16 in LDS (512x264 halves ~ 264KB: needs CDNA5 320KB LDS).
// Per step: g = [x_t | h] @ Wcomb via v_wmma_f32_16x16x32_f16 (8 K-chunks x 4 gates).
// Emits h sequence as f16 to yout (B,T,256), forward cols 0..127, backward 128..255.
// ---------------------------------------------------------------------------
__global__ __launch_bounds__(256) void lstm1_scan_kernel(
    const float* __restrict__ xin,
    const float* __restrict__ k1f, const float* __restrict__ r1f, const float* __restrict__ b1f,
    const float* __restrict__ k1b, const float* __restrict__ r1b, const float* __restrict__ b1b,
    _Float16* __restrict__ yout)
{
  __shared__ _Float16 WT[G1 * WP1];       // transposed [n][k] f16 weights
  __shared__ _Float16 xbuf[16 * AP1];     // current x_t tile (f16)
  __shared__ _Float16 hbuf[16 * AP1];     // current h tile (f16)

  const int dir  = blockIdx.y;
  const int bRow = blockIdx.x * 16;
  const int tid  = threadIdx.x;
  const float* Wk = dir ? k1b : k1f;
  const float* Wr = dir ? r1b : r1f;
  const float* bb = dir ? b1b : b1f;

  // Coalesced weight load: idx = k*512 + n  -> WT[n][k] (f16)
  for (int idx = tid; idx < G1 * K1; idx += 256) {
    const int n = idx & (G1 - 1);
    const int k = idx >> 9;
    const float w = (k < F_IN) ? Wk[idx] : Wr[idx - F_IN * G1];
    WT[n * WP1 + k] = (_Float16)w;
  }
  for (int idx = tid; idx < 16 * AP1; idx += 256) hbuf[idx] = (_Float16)0.0f;
  __syncthreads();

  const int wv   = tid >> 5;   // wave 0..7
  const int lane = tid & 31;
  const int colL = lane & 15;  // N column (C/D layout) == M row (A layout)
  const int hi   = lane >> 4;

  const float bi = bb[0 * H1 + wv * 16 + colL];
  const float bf = bb[1 * H1 + wv * 16 + colL];
  const float bc = bb[2 * H1 + wv * 16 + colL];
  const float bo = bb[3 * H1 + wv * 16 + colL];

  v8f c = {};   // cell state: column wv*16+colL, rows v + 8*hi

  for (int t = 0; t < T_SEQ; ++t) {
    const int tt = dir ? (T_SEQ - 1 - t) : t;

    // Stage x_t tile (16 x 128 f32 -> f16)
    for (int idx = tid; idx < 16 * F_IN; idx += 256) {
      const int r = idx >> 7, k = idx & (F_IN - 1);
      xbuf[r * AP1 + k] =
          (_Float16)xin[((size_t)(bRow + r) * T_SEQ + tt) * F_IN + k];
    }
    __syncthreads();  // also orders previous step's hbuf writes vs this step's reads

    v8f ai = {}, af = {}, ac = {}, ao = {};
#pragma unroll
    for (int kc = 0; kc < 8; ++kc) {
      const _Float16* ab = (kc < 4) ? xbuf : hbuf;
      const int kb = (kc & 3) * 32;
      // A fragment: lanes 0-15 K {0..7,16..23}, lanes 16-31 K {8..15,24..31}
      v16h a;
      {
        const _Float16* p = ab + colL * AP1 + kb + hi * 8;
        const v8h lo = *(const v8h*)p;
        const v8h hh = *(const v8h*)(p + 16);
#pragma unroll
        for (int j = 0; j < 8; ++j) { a[j] = lo[j]; a[8 + j] = hh[j]; }
      }
      const int kfull = kc * 32;
#pragma unroll
      for (int g = 0; g < 4; ++g) {
        const int n0 = g * H1 + wv * 16 + colL;
        const _Float16* p = WT + (size_t)n0 * WP1 + kfull + hi * 8;
        const v8h lo = *(const v8h*)p;
        const v8h hh = *(const v8h*)(p + 16);
        v16h bfrag;
#pragma unroll
        for (int j = 0; j < 8; ++j) { bfrag[j] = lo[j]; bfrag[8 + j] = hh[j]; }
        if (g == 0)
          ai = __builtin_amdgcn_wmma_f32_16x16x32_f16(false, a, false, bfrag, (short)0, ai, false, false);
        else if (g == 1)
          af = __builtin_amdgcn_wmma_f32_16x16x32_f16(false, a, false, bfrag, (short)0, af, false, false);
        else if (g == 2)
          ac = __builtin_amdgcn_wmma_f32_16x16x32_f16(false, a, false, bfrag, (short)0, ac, false, false);
        else
          ao = __builtin_amdgcn_wmma_f32_16x16x32_f16(false, a, false, bfrag, (short)0, ao, false, false);
      }
    }
    __syncthreads();  // all waves done reading hbuf/xbuf

    // Gate math + state update; write new h (f16) to LDS and to y sequence
#pragma unroll
    for (int v = 0; v < 8; ++v) {
      const int row = v + hi * 8;
      const float iv = sigmoidf_(ai[v] + bi);
      const float fv = sigmoidf_(af[v] + bf);
      float cc = ac[v] + bc; cc = cc > 0.f ? cc : 0.f;
      const float ov = sigmoidf_(ao[v] + bo);
      const float cv = fv * c[v] + iv * cc;
      c[v] = cv;
      const float rc = cv > 0.f ? cv : 0.f;
      const _Float16 hv = (_Float16)(ov * rc);
      hbuf[row * AP1 + wv * 16 + colL] = hv;
      yout[((size_t)(bRow + row) * T_SEQ + tt) * (2 * H1) + dir * H1 + wv * 16 + colL] = hv;
    }
  }
}

// ---------------------------------------------------------------------------
// Layer-2 LSTM scan. Input y (B,T,256) f16 read directly from global as
// A-fragments; recurrent part (K 256..319) from LDS h tile. 4 waves (H2=64).
// return_seq=False: only final h written (f32) to hout (B,128).
// ---------------------------------------------------------------------------
__global__ __launch_bounds__(128) void lstm2_scan_kernel(
    const _Float16* __restrict__ yin,
    const float* __restrict__ k2f, const float* __restrict__ r2f, const float* __restrict__ b2f,
    const float* __restrict__ k2b, const float* __restrict__ r2b, const float* __restrict__ b2b,
    float* __restrict__ hout)
{
  __shared__ _Float16 WT[G2 * WP2];     // 256x328 halves ~ 164KB
  __shared__ _Float16 hbuf[16 * AP2];

  const int dir  = blockIdx.y;
  const int bRow = blockIdx.x * 16;
  const int tid  = threadIdx.x;
  const float* Wk = dir ? k2b : k2f;
  const float* Wr = dir ? r2b : r2f;
  const float* bb = dir ? b2b : b2f;

  // idx = k*256 + n -> WT[n][k]
  for (int idx = tid; idx < G2 * K2; idx += 128) {
    const int n = idx & (G2 - 1);
    const int k = idx >> 8;
    const float w = (k < 256) ? Wk[idx] : Wr[idx - 256 * G2];
    WT[n * WP2 + k] = (_Float16)w;
  }
  for (int idx = tid; idx < 16 * AP2; idx += 128) hbuf[idx] = (_Float16)0.0f;
  __syncthreads();

  const int wv   = tid >> 5;   // 0..3
  const int lane = tid & 31;
  const int colL = lane & 15;
  const int hi   = lane >> 4;

  const float bi = bb[0 * H2 + wv * 16 + colL];
  const float bf = bb[1 * H2 + wv * 16 + colL];
  const float bc = bb[2 * H2 + wv * 16 + colL];
  const float bo = bb[3 * H2 + wv * 16 + colL];

  v8f c = {};
  v8f hreg = {};

  for (int t = 0; t < T_SEQ; ++t) {
    const int tt = dir ? (T_SEQ - 1 - t) : t;

    v8f ai = {}, af = {}, ac = {}, ao = {};
#pragma unroll
    for (int kc = 0; kc < 10; ++kc) {
      v8h lo, hh;
      if (kc < 8) {   // x part, straight from global f16 sequence
        const _Float16* p =
            yin + ((size_t)(bRow + colL) * T_SEQ + tt) * 256 + kc * 32 + hi * 8;
        lo = *(const v8h*)p;
        hh = *(const v8h*)(p + 16);
      } else {        // recurrent part from LDS
        const _Float16* p = hbuf + colL * AP2 + (kc - 8) * 32 + hi * 8;
        lo = *(const v8h*)p;
        hh = *(const v8h*)(p + 16);
      }
      v16h a;
#pragma unroll
      for (int j = 0; j < 8; ++j) { a[j] = lo[j]; a[8 + j] = hh[j]; }

      const int kfull = kc * 32;
#pragma unroll
      for (int g = 0; g < 4; ++g) {
        const int n0 = g * H2 + wv * 16 + colL;
        const _Float16* p = WT + (size_t)n0 * WP2 + kfull + hi * 8;
        const v8h blo = *(const v8h*)p;
        const v8h bhi = *(const v8h*)(p + 16);
        v16h bfrag;
#pragma unroll
        for (int j = 0; j < 8; ++j) { bfrag[j] = blo[j]; bfrag[8 + j] = bhi[j]; }
        if (g == 0)
          ai = __builtin_amdgcn_wmma_f32_16x16x32_f16(false, a, false, bfrag, (short)0, ai, false, false);
        else if (g == 1)
          af = __builtin_amdgcn_wmma_f32_16x16x32_f16(false, a, false, bfrag, (short)0, af, false, false);
        else if (g == 2)
          ac = __builtin_amdgcn_wmma_f32_16x16x32_f16(false, a, false, bfrag, (short)0, ac, false, false);
        else
          ao = __builtin_amdgcn_wmma_f32_16x16x32_f16(false, a, false, bfrag, (short)0, ao, false, false);
      }
    }
    __syncthreads();  // hbuf reads complete

#pragma unroll
    for (int v = 0; v < 8; ++v) {
      const int row = v + hi * 8;
      const float iv = sigmoidf_(ai[v] + bi);
      const float fv = sigmoidf_(af[v] + bf);
      float cc = ac[v] + bc; cc = cc > 0.f ? cc : 0.f;
      const float ov = sigmoidf_(ao[v] + bo);
      const float cv = fv * c[v] + iv * cc;
      c[v] = cv;
      const float rc = cv > 0.f ? cv : 0.f;
      const float hv = ov * rc;
      hreg[v] = hv;
      hbuf[row * AP2 + wv * 16 + colL] = (_Float16)hv;
    }
    __syncthreads();  // hbuf writes visible before next step's reads
  }

#pragma unroll
  for (int v = 0; v < 8; ++v) {
    const int row = v + hi * 8;
    hout[(size_t)(bRow + row) * 128 + dir * 64 + wv * 16 + colL] = hreg[v];
  }
}

// ---------------------------------------------------------------------------
// Final heads: combined(192) -> relu(·@w(192,2)+b)@w(2,1)+b for z and t.
// ---------------------------------------------------------------------------
__global__ __launch_bounds__(256) void heads_kernel(
    const float* __restrict__ x64, const float* __restrict__ hcat,
    const float* __restrict__ wz1, const float* __restrict__ bz1,
    const float* __restrict__ wz2, const float* __restrict__ bz2,
    const float* __restrict__ wt1, const float* __restrict__ bt1,
    const float* __restrict__ wt2, const float* __restrict__ bt2,
    float* __restrict__ out)
{
  const int b = blockIdx.x * blockDim.x + threadIdx.x;
  if (b >= BATCH) return;
  float comb[192];
#pragma unroll 8
  for (int j = 0; j < 64; ++j) comb[j] = x64[b * 64 + j];
#pragma unroll 8
  for (int j = 0; j < 128; ++j) comb[64 + j] = hcat[b * 128 + j];

  {
    float h0 = bz1[0], h1v = bz1[1];
    for (int k = 0; k < 192; ++k) { h0 += comb[k] * wz1[k * 2]; h1v += comb[k] * wz1[k * 2 + 1]; }
    h0 = h0 > 0.f ? h0 : 0.f; h1v = h1v > 0.f ? h1v : 0.f;
    out[b] = h0 * wz2[0] + h1v * wz2[1] + bz2[0];
  }
  {
    float h0 = bt1[0], h1v = bt1[1];
    for (int k = 0; k < 192; ++k) { h0 += comb[k] * wt1[k * 2]; h1v += comb[k] * wt1[k * 2 + 1]; }
    h0 = h0 > 0.f ? h0 : 0.f; h1v = h1v > 0.f ? h1v : 0.f;
    out[BATCH + b] = h0 * wt2[0] + h1v * wt2[1] + bt2[0];
  }
}

// ---------------------------------------------------------------------------
extern "C" void kernel_launch(void* const* d_in, const int* in_sizes, int n_in,
                              void* d_out, int out_size, void* d_ws, size_t ws_size,
                              hipStream_t stream) {
  const int*   inputA = (const int*)d_in[0];
  const float* inputB = (const float*)d_in[1];
  const float* emb    = (const float*)d_in[2];
  const float* w1 = (const float*)d_in[3];  const float* b1 = (const float*)d_in[4];
  const float* w2 = (const float*)d_in[5];  const float* b2 = (const float*)d_in[6];
  const float* w3 = (const float*)d_in[7];  const float* b3 = (const float*)d_in[8];
  const float* k1f = (const float*)d_in[9];  const float* r1f = (const float*)d_in[10]; const float* bb1f = (const float*)d_in[11];
  const float* k1b = (const float*)d_in[12]; const float* r1b = (const float*)d_in[13]; const float* bb1b = (const float*)d_in[14];
  const float* k2f = (const float*)d_in[15]; const float* r2f = (const float*)d_in[16]; const float* bb2f = (const float*)d_in[17];
  const float* k2b = (const float*)d_in[18]; const float* r2b = (const float*)d_in[19]; const float* bb2b = (const float*)d_in[20];
  const float* wz1 = (const float*)d_in[21]; const float* bz1 = (const float*)d_in[22];
  const float* wz2 = (const float*)d_in[23]; const float* bz2 = (const float*)d_in[24];
  const float* wt1 = (const float*)d_in[25]; const float* bt1 = (const float*)d_in[26];
  const float* wt2 = (const float*)d_in[27]; const float* bt2 = (const float*)d_in[28];
  float* out = (float*)d_out;

  // Workspace: x64 (B*64 f32) | y1 (B*T*256 f16) | hcat (B*128 f32)  ~= 64.2 MB
  char* ws = (char*)d_ws;
  float*    x64  = (float*)ws;                                   // 65536 B
  _Float16* y1   = (_Float16*)(ws + 65536);                      // 67108864 B
  float*    hcat = (float*)(ws + 65536 + (size_t)BATCH * T_SEQ * 256 * 2);

  emb_mlp_kernel<<<BATCH, 256, 0, stream>>>(inputA, emb, w1, b1, w2, b2, w3, b3, x64);

  dim3 g1(BATCH / 16, 2);
  lstm1_scan_kernel<<<g1, 256, 0, stream>>>(inputB, k1f, r1f, bb1f, k1b, r1b, bb1b, y1);

  dim3 g2(BATCH / 16, 2);
  lstm2_scan_kernel<<<g2, 128, 0, stream>>>(y1, k2f, r2f, bb2f, k2b, r2b, bb2b, hcat);

  heads_kernel<<<1, 256, 0, stream>>>(x64, hcat, wz1, bz1, wz2, bz2, wt1, bt1, wt2, bt2, out);
}
